// KG_GAT_Adapter_60936995995912
// MI455X (gfx1250) — matrix operand order repaired and checked by reading, not compiled, not run on hardware
//
#include <hip/hip_runtime.h>
#include <hip/hip_bf16.h>

// ---------------- problem constants ----------------
#define NN      10000
#define IN_DIM  4096
#define HID     1024
#define OUT_DIM 4096
#define HEADS   4
#define HC      4096      // HEADS * HID
#define NE      100000
#define NET     110000    // NE + NN self loops

typedef __attribute__((ext_vector_type(16))) __bf16 v16bf;
typedef __attribute__((ext_vector_type(8)))  float  v8f;

union Frag { v16bf v; uint4 q[2]; };

// ---------------- fp32 -> bf16 (RNE), row-major ----------------
__global__ __launch_bounds__(256) void cvt_bf16(const float* __restrict__ in,
                                                unsigned short* __restrict__ out, int n) {
  int i = blockIdx.x * 256 + threadIdx.x;
  if (i < n) {
    unsigned int u = __float_as_uint(in[i]);
    u += 0x7FFFu + ((u >> 16) & 1u);
    out[i] = (unsigned short)(u >> 16);
  }
}

// ---------------- fp32 [K,N=4096] -> bf16 transposed [N,K] ----------------
__global__ __launch_bounds__(256) void cvt_bf16_t(const float* __restrict__ in,
                                                  unsigned short* __restrict__ out, int K) {
  long long idx = (long long)blockIdx.x * 256 + threadIdx.x;
  if (idx >= (long long)K * 4096) return;
  int k = (int)(idx >> 12);
  int n = (int)(idx & 4095);
  unsigned int u = __float_as_uint(in[idx]);
  u += 0x7FFFu + ((u >> 16) & 1u);
  out[(size_t)n * K + k] = (unsigned short)(u >> 16);
}

// ---------------- WMMA bf16 GEMM: C[M,N] = A[M,K] @ BT[N,K]^T (+bias) ----------------
// Block: 256 threads = 8 waves; tile 128(M) x 128(N), BK=32.
// Wave (wm 0..3, wn 0..1) owns 32(M) x 64(N) = 2x4 wmma tiles -> 8 v_wmma per K-step.
// Double-buffered LDS staged via global_load_async_to_lds_b128:
//   prefetch tile k+1 into buffer cur^1, then s_wait_asynccnt 4 so only the
//   current buffer's 4 loads are waited on -> async engine overlaps WMMA pipe.
__global__ __launch_bounds__(256) void gemm_bf16_wmma(
    const unsigned short* __restrict__ A,    // [M,K] bf16 bits row-major
    const unsigned short* __restrict__ BT,   // [N,K] bf16 bits row-major (pre-transposed W)
    float* __restrict__ C,                   // [M,N] fp32
    const float* __restrict__ bias,          // [N] or nullptr
    int M, int N, int K) {
  __shared__ __align__(16) unsigned short As[2][128][32];   // 16 KB
  __shared__ __align__(16) unsigned short Bs[2][128][32];   // 16 KB, [n][k]

  const int tid   = threadIdx.x;
  const int wave  = tid >> 5, lane = tid & 31;
  const int wm    = wave & 3, wn = wave >> 2;
  const int khalf = lane >> 4, l16 = lane & 15;
  const int blockM = blockIdx.y * 128, blockN = blockIdx.x * 128;

  // staging assignment: chunk c (0..511 per tile) -> row = c>>2, k8 = (c&3)*8
  const int m0 = tid >> 2, k8 = (tid & 3) * 8;      // chunks tid and tid+256
  const int m1 = (tid + 256) >> 2;
  int gm0 = blockM + m0; gm0 = (gm0 < M) ? gm0 : (M - 1);
  int gm1 = blockM + m1; gm1 = (gm1 < M) ? gm1 : (M - 1);
  const int gn0 = blockN + m0, gn1 = blockN + m1;   // N always multiple of 128

  const unsigned ldsA0 = (unsigned)(size_t)&As[0][m0][k8];
  const unsigned ldsA1 = (unsigned)(size_t)&As[0][m1][k8];
  const unsigned ldsB0 = (unsigned)(size_t)&Bs[0][m0][k8];
  const unsigned ldsB1 = (unsigned)(size_t)&Bs[0][m1][k8];

  const unsigned short* gA0 = A  + (size_t)gm0 * K + k8;
  const unsigned short* gA1 = A  + (size_t)gm1 * K + k8;
  const unsigned short* gB0 = BT + (size_t)gn0 * K + k8;
  const unsigned short* gB1 = BT + (size_t)gn1 * K + k8;

  auto stage = [&](int kt, int buf) {
    const unsigned o = (unsigned)buf * 8192u;
    unsigned long long ga0 = (unsigned long long)(size_t)(gA0 + kt);
    unsigned long long ga1 = (unsigned long long)(size_t)(gA1 + kt);
    unsigned long long gb0 = (unsigned long long)(size_t)(gB0 + kt);
    unsigned long long gb1 = (unsigned long long)(size_t)(gB1 + kt);
    asm volatile("global_load_async_to_lds_b128 %0, %1, off" :: "v"(ldsA0 + o), "v"(ga0) : "memory");
    asm volatile("global_load_async_to_lds_b128 %0, %1, off" :: "v"(ldsA1 + o), "v"(ga1) : "memory");
    asm volatile("global_load_async_to_lds_b128 %0, %1, off" :: "v"(ldsB0 + o), "v"(gb0) : "memory");
    asm volatile("global_load_async_to_lds_b128 %0, %1, off" :: "v"(ldsB1 + o), "v"(gb1) : "memory");
  };

  v8f acc[2][4];
  #pragma unroll
  for (int a = 0; a < 2; ++a)
    #pragma unroll
    for (int b = 0; b < 4; ++b) acc[a][b] = (v8f)0.0f;

  stage(0, 0);                                 // prologue: fill buffer 0
  int cur = 0;

  for (int kt = 0; kt < K; kt += 32) {
    // prefetch next K-tile into the other buffer (redundant re-fetch on last iter)
    int ktn = (kt + 32 < K) ? (kt + 32) : kt;
    stage(ktn, cur ^ 1);
    // wait only for the current buffer's 4 loads (next tile's 4 stay in flight)
    asm volatile("s_wait_asynccnt 0x4" ::: "memory");
    __syncthreads();

    // ---- build fragments (all 16B ds_load_b128) ----
    const unsigned short (*Ac)[32] = As[cur];
    const unsigned short (*Bc)[32] = Bs[cur];
    Frag af[2], bf[4];
    #pragma unroll
    for (int t = 0; t < 2; ++t) {
      int mrow = wm * 32 + t * 16 + l16;
      af[t].q[0] = *(const uint4*)&Ac[mrow][khalf * 8];        // K = khalf*8 + 0..7
      af[t].q[1] = *(const uint4*)&Ac[mrow][16 + khalf * 8];   // K = 16+khalf*8 + 0..7
    }
    #pragma unroll
    for (int t = 0; t < 4; ++t) {
      int ncol = wn * 64 + t * 16 + l16;
      bf[t].q[0] = *(const uint4*)&Bc[ncol][khalf * 8];        // K = khalf*8 + 0..7
      bf[t].q[1] = *(const uint4*)&Bc[ncol][16 + khalf * 8];   // K = 16+khalf*8 + 0..7
    }

    #pragma unroll
    for (int tm = 0; tm < 2; ++tm)
      #pragma unroll
      for (int tn = 0; tn < 4; ++tn)
        acc[tm][tn] = __builtin_amdgcn_wmma_f32_16x16x32_bf16(
            false, af[tm].v, false, bf[tn].v, (short)0, acc[tm][tn], false, false);

    __syncthreads();   // all waves done reading `cur` before it is overwritten
    cur ^= 1;
  }

  // ---- store: VGPR r -> row = base + khalf*8 + r, col = base + l16 ----
  #pragma unroll
  for (int tm = 0; tm < 2; ++tm) {
    #pragma unroll
    for (int tn = 0; tn < 4; ++tn) {
      int col = blockN + wn * 64 + tn * 16 + l16;
      float bv = bias ? bias[col] : 0.0f;
      #pragma unroll
      for (int r = 0; r < 8; ++r) {
        int row = blockM + wm * 32 + tm * 16 + khalf * 8 + r;
        if (row < M) C[(size_t)row * N + col] = acc[tm][tn][r] + bv;
      }
    }
  }
}

// ---------------- attention coefficients: one wave per (node, head) ----------------
__global__ __launch_bounds__(256) void attn_coef(const float* __restrict__ h,
                                                 const float* __restrict__ attS,
                                                 const float* __restrict__ attD,
                                                 float* __restrict__ aS,
                                                 float* __restrict__ aD) {
  int gw   = (blockIdx.x * 256 + threadIdx.x) >> 5;
  int lane = threadIdx.x & 31;
  if (gw >= NN * HEADS) return;
  int n = gw >> 2, hd = gw & 3;
  const float* hp = h + (size_t)n * HC + hd * HID;
  const float* ps = attS + hd * HID;
  const float* pd = attD + hd * HID;
  float s0 = 0.f, s1 = 0.f;
  for (int c = lane; c < HID; c += 32) {
    float v = hp[c];
    s0 += v * ps[c];
    s1 += v * pd[c];
  }
  #pragma unroll
  for (int o = 16; o; o >>= 1) {
    s0 += __shfl_xor(s0, o, 32);
    s1 += __shfl_xor(s1, o, 32);
  }
  if (lane == 0) { aS[gw] = s0; aD[gw] = s1; }
}

// ---------------- softmax helpers ----------------
__global__ __launch_bounds__(256) void init_maxdenom(float* __restrict__ amax,
                                                     float* __restrict__ denom, int n) {
  int i = blockIdx.x * 256 + threadIdx.x;
  if (i < n) { amax[i] = -3.0e38f; denom[i] = 0.f; }
}

__device__ __forceinline__ void atomicMaxF(float* addr, float val) {
  if (val >= 0.f) atomicMax((int*)addr, __float_as_int(val));
  else            atomicMin((unsigned int*)addr, __float_as_uint(val));
}

__global__ __launch_bounds__(256) void edge_max(const int* __restrict__ ei,
                                                const float* __restrict__ aS,
                                                const float* __restrict__ aD,
                                                float* __restrict__ alpha,
                                                float* __restrict__ amax) {
  int e = blockIdx.x * 256 + threadIdx.x;
  if (e >= NET) return;
  int src = (e < NE) ? ei[e]      : (e - NE);
  int dst = (e < NE) ? ei[NE + e] : (e - NE);
  #pragma unroll
  for (int hd = 0; hd < HEADS; ++hd) {
    float a = aS[src * HEADS + hd] + aD[dst * HEADS + hd];
    a = (a > 0.f) ? a : 0.2f * a;                 // leaky_relu
    alpha[(size_t)e * HEADS + hd] = a;
    atomicMaxF(&amax[dst * HEADS + hd], a);
  }
}

__global__ __launch_bounds__(256) void edge_expsum(const int* __restrict__ ei,
                                                   float* __restrict__ alpha,
                                                   const float* __restrict__ amax,
                                                   float* __restrict__ denom) {
  int e = blockIdx.x * 256 + threadIdx.x;
  if (e >= NET) return;
  int dst = (e < NE) ? ei[NE + e] : (e - NE);
  #pragma unroll
  for (int hd = 0; hd < HEADS; ++hd) {
    float v = __expf(alpha[(size_t)e * HEADS + hd] - amax[dst * HEADS + hd]);
    alpha[(size_t)e * HEADS + hd] = v;
    atomicAdd(&denom[dst * HEADS + hd], v);
  }
}

// ---------------- weighted scatter-add: one block per edge ----------------
__global__ __launch_bounds__(256) void edge_aggregate(const int* __restrict__ ei,
                                                      const float* __restrict__ h,
                                                      const float* __restrict__ alpha,
                                                      const float* __restrict__ denom,
                                                      float* __restrict__ agg) {
  int e = blockIdx.x;
  int src = (e < NE) ? ei[e]      : (e - NE);
  int dst = (e < NE) ? ei[NE + e] : (e - NE);
  __shared__ float w[HEADS];
  if (threadIdx.x < HEADS)
    w[threadIdx.x] = alpha[(size_t)e * HEADS + threadIdx.x] / denom[dst * HEADS + threadIdx.x];
  __syncthreads();
  const float* hs = h   + (size_t)src * HC;
  float*       ag = agg + (size_t)dst * HC;
  for (int i = threadIdx.x; i < HC; i += 256)
    atomicAdd(&ag[i], hs[i] * w[i >> 10]);        // head = i / HID
}

// ---------------- head mean + bias + ELU ----------------
__global__ __launch_bounds__(256) void head_mean_bias_elu(const float* __restrict__ agg,
                                                          const float* __restrict__ b,
                                                          float* __restrict__ feat) {
  int idx = blockIdx.x * 256 + threadIdx.x;
  if (idx >= NN * HID) return;
  int n = idx >> 10, c = idx & 1023;
  float s = 0.f;
  #pragma unroll
  for (int hd = 0; hd < HEADS; ++hd) s += agg[(size_t)n * HC + hd * HID + c];
  float v = s * 0.25f + b[c];
  feat[idx] = (v > 0.f) ? v : (__expf(v) - 1.0f);
}

// ---------------- fused LayerNorm + L2 normalize (block per row) ----------------
__device__ __forceinline__ float blockReduceSum(float v, float* sred) {
  #pragma unroll
  for (int o = 16; o; o >>= 1) v += __shfl_xor(v, o, 32);
  int wave = threadIdx.x >> 5, lane = threadIdx.x & 31;
  __syncthreads();
  if (lane == 0) sred[wave] = v;
  __syncthreads();
  float t = 0.f;
  #pragma unroll
  for (int w = 0; w < 8; ++w) t += sred[w];
  return t;
}

__global__ __launch_bounds__(256) void ln_norm(float* __restrict__ y,
                                               const float* __restrict__ gamma,
                                               const float* __restrict__ beta) {
  __shared__ float sred[8];
  float* p = y + (size_t)blockIdx.x * OUT_DIM;
  int tid = threadIdx.x;
  float s = 0.f;
  for (int i = tid; i < OUT_DIM; i += 256) s += p[i];
  s = blockReduceSum(s, sred);
  float mu = s * (1.0f / OUT_DIM);
  float v = 0.f;
  for (int i = tid; i < OUT_DIM; i += 256) { float d = p[i] - mu; v += d * d; }
  v = blockReduceSum(v, sred);
  float rstd = rsqrtf(v * (1.0f / OUT_DIM) + 1e-5f);
  float ss = 0.f;
  for (int i = tid; i < OUT_DIM; i += 256) {
    float z = (p[i] - mu) * rstd * gamma[i] + beta[i];
    p[i] = z;
    ss += z * z;
  }
  ss = blockReduceSum(ss, sred);
  float inv = 1.0f / fmaxf(sqrtf(ss), 1e-12f);
  for (int i = tid; i < OUT_DIM; i += 256) p[i] *= inv;
}

// ==================================================================
extern "C" void kernel_launch(void* const* d_in, const int* in_sizes, int n_in,
                              void* d_out, int out_size, void* d_ws, size_t ws_size,
                              hipStream_t stream) {
  const float* x     = (const float*)d_in[0];
  const int*   ei    = (const int*)  d_in[1];
  const float* W1    = (const float*)d_in[2];
  const float* a1s   = (const float*)d_in[3];
  const float* a1d   = (const float*)d_in[4];
  const float* b1    = (const float*)d_in[5];
  const float* W2    = (const float*)d_in[6];
  const float* a2s   = (const float*)d_in[7];
  const float* a2d   = (const float*)d_in[8];
  const float* b2    = (const float*)d_in[9];
  const float* Wp    = (const float*)d_in[10];
  const float* bp    = (const float*)d_in[11];
  const float* gamma = (const float*)d_in[12];
  const float* beta  = (const float*)d_in[13];
  float* out = (float*)d_out;

  // ---- workspace carve-out ----
  char* ws = (char*)d_ws;
  size_t off = 0;
  auto take = [&](size_t bytes) -> void* {
    void* p = ws + off;
    off += (bytes + 255) & ~(size_t)255;
    return p;
  };
  unsigned short* xb   = (unsigned short*)take((size_t)NN * IN_DIM * 2);   // activations bf16 [M,K]
  unsigned short* wbuf = (unsigned short*)take((size_t)IN_DIM * HC * 2);   // weights bf16 [N,K] (transposed)
  float* hbuf  = (float*)take((size_t)NN * HC * 4);                        // h = x@W
  float* agg   = (float*)take((size_t)NN * HC * 4);                        // scatter accumulator
  float* feat  = (float*)take((size_t)NN * HID * 4);                       // post-ELU features
  float* aSrc  = (float*)take((size_t)NN * HEADS * 4);
  float* aDst  = (float*)take((size_t)NN * HEADS * 4);
  float* amax  = (float*)take((size_t)NN * HEADS * 4);
  float* denom = (float*)take((size_t)NN * HEADS * 4);
  float* alpha = (float*)take((size_t)NET * HEADS * 4);

  const dim3 blk(256);
  auto cdiv = [](long long a, long long b) { return (unsigned)((a + b - 1) / b); };

  auto run_gat = [&](const float* xin, int Kin, const float* W,
                     const float* attS, const float* attD, const float* b) {
    // 1. convert: activations row-major bf16, weights transposed bf16 [N,K]
    cvt_bf16  <<<cdiv((long long)NN * Kin, 256), blk, 0, stream>>>(xin, xb, NN * Kin);
    cvt_bf16_t<<<cdiv((long long)Kin * HC, 256), blk, 0, stream>>>(W, wbuf, Kin);
    // 2. h = x @ W  (WMMA bf16, double-buffered async-LDS staging)
    dim3 ggrid(HC / 128, cdiv(NN, 128));
    gemm_bf16_wmma<<<ggrid, blk, 0, stream>>>(xb, wbuf, hbuf, nullptr, NN, HC, Kin);
    // 3. attention logits
    attn_coef<<<cdiv((long long)NN * HEADS * 32, 256), blk, 0, stream>>>(hbuf, attS, attD, aSrc, aDst);
    // 4. segment softmax over incoming edges
    init_maxdenom<<<cdiv(NN * HEADS, 256), blk, 0, stream>>>(amax, denom, NN * HEADS);
    edge_max   <<<cdiv(NET, 256), blk, 0, stream>>>(ei, aSrc, aDst, alpha, amax);
    edge_expsum<<<cdiv(NET, 256), blk, 0, stream>>>(ei, alpha, amax, denom);
    // 5. weighted scatter-add aggregation
    hipMemsetAsync(agg, 0, (size_t)NN * HC * 4, stream);
    edge_aggregate<<<NET, blk, 0, stream>>>(ei, hbuf, alpha, denom, agg);
    // 6. head mean + bias + ELU
    head_mean_bias_elu<<<cdiv((long long)NN * HID, 256), blk, 0, stream>>>(agg, b, feat);
  };

  run_gat(x,    IN_DIM, W1, a1s, a1d, b1);   // layer 1: feat <- elu(GAT(x))
  run_gat(feat, HID,    W2, a2s, a2d, b2);   // layer 2: feat <- elu(GAT(feat))

  // projection: out = feat @ Wp + bp  (WMMA bf16)
  cvt_bf16  <<<cdiv((long long)NN * HID, 256), blk, 0, stream>>>(feat, xb, NN * HID);
  cvt_bf16_t<<<cdiv((long long)HID * OUT_DIM, 256), blk, 0, stream>>>(Wp, wbuf, HID);
  dim3 pgrid(OUT_DIM / 128, cdiv(NN, 128));
  gemm_bf16_wmma<<<pgrid, blk, 0, stream>>>(xb, wbuf, out, bp, NN, OUT_DIM, HID);

  // LayerNorm + F.normalize, fused, in-place on d_out
  ln_norm<<<NN, blk, 0, stream>>>(out, gamma, beta);
}
